// SampleNetDD_21964462752419
// MI455X (gfx1250) — matrix operand order repaired and checked by reading, not compiled, not run on hardware
//
#include <hip/hip_runtime.h>
#include <hip/hip_bf16.h>

#define BB   16
#define PP   1024
#define KK   20
#define LL   10
#define NPTS (BB * PP)          // 16384
#define RR   (NPTS * KK)        // 327680
#define NCLS 40

typedef __attribute__((ext_vector_type(16))) _Float16 v16h;
typedef __attribute__((ext_vector_type(8)))  float    v8f;
typedef __attribute__((ext_vector_type(4)))  unsigned int u32x4;

union F16Frag { u32x4 u[2]; v16h v; };

// ---------------------------------------------------------------------------
// A fragment: 16x32 fp16, row-major source. Lane layout per CDNA5 ISA 7.12.2:
// lanes 0-15 hold row m=lane, VGPR0-3 = K[lh*8 .. lh*8+7], VGPR4-7 = K[16+lh*8 ..]
// => two contiguous 16B loads per lane.
__device__ inline v16h frag_a(const _Float16* __restrict__ rowPtr, int kBase, int lh) {
    F16Frag f;
    f.u[0] = *reinterpret_cast<const u32x4*>(rowPtr + kBase + lh * 8);
    f.u[1] = *reinterpret_cast<const u32x4*>(rowPtr + kBase + 16 + lh * 8);
    return f.v;
}

// B fragment: pre-packed so lane's 8 dwords are contiguous (32B).
__device__ inline v16h frag_b(const unsigned int* __restrict__ tileBase, int lane) {
    const u32x4* p = reinterpret_cast<const u32x4*>(tileBase + lane * 8);
    F16Frag f;
    f.u[0] = p[0];
    f.u[1] = p[1];
    return f.v;
}

// ---------------------------------------------------------------------------
// Kernel 1: brute-force kNN (K=20) per cloud, LDS-staged positions.
__global__ void __launch_bounds__(256) knn_kernel(const float* __restrict__ pos,
                                                  int* __restrict__ nbr) {
    __shared__ float sx[PP], sy[PP], sz[PP];
    const int b = blockIdx.x >> 2;
    const int cbase = (blockIdx.x & 3) * 256;
    const float* pb = pos + (size_t)b * PP * 3;
    for (int i = threadIdx.x; i < PP; i += 256) {
        sx[i] = pb[i * 3 + 0];
        sy[i] = pb[i * 3 + 1];
        sz[i] = pb[i * 3 + 2];
    }
    __syncthreads();
    const int c = cbase + threadIdx.x;
    const float cx = sx[c], cy = sy[c], cz = sz[c];
    float dist[KK];
    int   idx[KK];
#pragma unroll
    for (int t = 0; t < KK; ++t) { dist[t] = 1e30f; idx[t] = 0; }
    for (int j = 0; j < PP; ++j) {
        float dx = sx[j] - cx, dy = sy[j] - cy, dz = sz[j] - cz;
        float d2 = dx * dx + dy * dy + dz * dz;
        if (j == c) continue;
        if (d2 < dist[KK - 1]) {
            dist[KK - 1] = d2; idx[KK - 1] = j;
#pragma unroll
            for (int t = KK - 1; t > 0; --t) {
                if (dist[t] < dist[t - 1]) {
                    float td = dist[t]; dist[t] = dist[t - 1]; dist[t - 1] = td;
                    int   ti = idx[t];  idx[t]  = idx[t - 1];  idx[t - 1]  = ti;
                }
            }
        }
    }
    const int g = b * PP + c;
#pragma unroll
    for (int t = 0; t < KK; ++t) nbr[(size_t)g * KK + t] = b * PP + idx[t];
}

// ---------------------------------------------------------------------------
// Kernel 2: covariance + Jacobi(5) + dir_clusters + sign + h/feat MLP.
__global__ void __launch_bounds__(256) geom_kernel(const float* __restrict__ pos,
                                                   const int* __restrict__ nbr,
                                                   const float* __restrict__ W1,
                                                   const float* __restrict__ b1,
                                                   const float* __restrict__ W2,
                                                   const float* __restrict__ b2,
                                                   float* __restrict__ Vws,
                                                   float* __restrict__ featws,
                                                   float* __restrict__ dirws) {
    __shared__ float sW1[96], sb1[32], sW2[480], sb2[15];
    for (int i = threadIdx.x; i < 96;  i += 256) sW1[i] = W1[i];
    for (int i = threadIdx.x; i < 32;  i += 256) sb1[i] = b1[i];
    for (int i = threadIdx.x; i < 480; i += 256) sW2[i] = W2[i];
    for (int i = threadIdx.x; i < 15;  i += 256) sb2[i] = b2[i];
    __syncthreads();

    const int n = blockIdx.x * 256 + threadIdx.x;
    const float px = pos[n * 3 + 0], py = pos[n * 3 + 1], pz = pos[n * 3 + 2];

    float cl[KK][3];
#pragma unroll
    for (int k = 0; k < KK; ++k) {
        int j = nbr[(size_t)n * KK + k];
        cl[k][0] = pos[j * 3 + 0] - px;
        cl[k][1] = pos[j * 3 + 1] - py;
        cl[k][2] = pos[j * 3 + 2] - pz;
    }

    float A[3][3] = {{0.f,0.f,0.f},{0.f,0.f,0.f},{0.f,0.f,0.f}};
#pragma unroll
    for (int k = 0; k < LL; ++k)
#pragma unroll
        for (int i = 0; i < 3; ++i)
#pragma unroll
            for (int j = 0; j < 3; ++j) A[i][j] += cl[k][i] * cl[k][j];

    float V[3][3] = {{1.f,0.f,0.f},{0.f,1.f,0.f},{0.f,0.f,1.f}};
    const int PQp[3] = {0, 0, 1};
    const int PQq[3] = {1, 2, 2};
#pragma unroll
    for (int it = 0; it < 5; ++it) {
#pragma unroll
        for (int r = 0; r < 3; ++r) {
            const int p = PQp[r], q = PQq[r];
            float app = A[p][p], aqq = A[q][q], apq = A[p][q];
            bool  nz  = fabsf(apq) > 1e-12f;
            float den = nz ? 2.f * apq : 1.f;
            float tau = (aqq - app) / den;
            float sgn = (tau > 0.f) ? 1.f : ((tau < 0.f) ? -1.f : 0.f);
            float t   = nz ? (sgn / (fabsf(tau) + sqrtf(1.f + tau * tau))) : 0.f;
            float cc  = 1.f / sqrtf(1.f + t * t);
            float ss  = t * cc;
            float G[3][3] = {{1.f,0.f,0.f},{0.f,1.f,0.f},{0.f,0.f,1.f}};
            G[p][p] = cc; G[q][q] = cc; G[p][q] = ss; G[q][p] = -ss;
            float M[3][3];
#pragma unroll
            for (int i = 0; i < 3; ++i)
#pragma unroll
                for (int j = 0; j < 3; ++j)
                    M[i][j] = A[i][0]*G[0][j] + A[i][1]*G[1][j] + A[i][2]*G[2][j];
#pragma unroll
            for (int i = 0; i < 3; ++i)
#pragma unroll
                for (int j = 0; j < 3; ++j)
                    A[i][j] = G[0][i]*M[0][j] + G[1][i]*M[1][j] + G[2][i]*M[2][j];
            float Vn[3][3];
#pragma unroll
            for (int i = 0; i < 3; ++i)
#pragma unroll
                for (int j = 0; j < 3; ++j)
                    Vn[i][j] = V[i][0]*G[0][j] + V[i][1]*G[1][j] + V[i][2]*G[2][j];
#pragma unroll
            for (int i = 0; i < 3; ++i)
#pragma unroll
                for (int j = 0; j < 3; ++j) V[i][j] = Vn[i][j];
        }
    }

    // dir = clusters @ V (accumulated rotation); z-sign flip
    float dirz[KK];
    float zsum = 0.f;
#pragma unroll
    for (int k = 0; k < KK; ++k) {
        dirz[k] = cl[k][0]*V[0][2] + cl[k][1]*V[1][2] + cl[k][2]*V[2][2];
        zsum += dirz[k];
    }
    const float sg = (zsum > 0.f) ? 1.f : ((zsum < 0.f) ? -1.f : 0.f);

    float hs[32];
#pragma unroll
    for (int c = 0; c < 32; ++c) hs[c] = 0.f;
#pragma unroll
    for (int k = 0; k < KK; ++k) {
        float d0 = cl[k][0]*V[0][0] + cl[k][1]*V[1][0] + cl[k][2]*V[2][0];
        float d1 = cl[k][0]*V[0][1] + cl[k][1]*V[1][1] + cl[k][2]*V[2][1];
        float d2 = dirz[k] * sg;
        size_t r = (size_t)n * KK + k;
        dirws[r * 4 + 0] = d0;
        dirws[r * 4 + 1] = d1;
        dirws[r * 4 + 2] = d2;
        dirws[r * 4 + 3] = 0.f;
#pragma unroll
        for (int c = 0; c < 32; ++c) {
            float pre = d0 * sW1[c] + d1 * sW1[32 + c] + d2 * sW1[64 + c] + sb1[c];
            hs[c] += fmaxf(pre, 0.f);
        }
    }
#pragma unroll
    for (int o = 0; o < 15; ++o) {
        float acc = (float)KK * sb2[o];
#pragma unroll
        for (int c = 0; c < 32; ++c) acc += hs[c] * sW2[c * 15 + o];
        featws[(size_t)n * 16 + o] = 1.f / (1.f + expf(-acc));
    }
    featws[(size_t)n * 16 + 15] = 0.f;
#pragma unroll
    for (int i = 0; i < 3; ++i)
#pragma unroll
        for (int j = 0; j < 3; ++j) Vws[(size_t)n * 9 + i * 3 + j] = V[i][j];
}

// ---------------------------------------------------------------------------
// Kernel 3: build concat [R x 32] fp16 (18 real + 14 zero pad).
__global__ void __launch_bounds__(256) concat_kernel(const int* __restrict__ nbr,
                                                     const float* __restrict__ Vws,
                                                     const float* __restrict__ featws,
                                                     const float* __restrict__ dirws,
                                                     _Float16* __restrict__ X) {
    const size_t r = (size_t)blockIdx.x * 256 + threadIdx.x;
    const int n = (int)(r / KK);
    const float* V = Vws + (size_t)n * 9;
    float out[18];
    out[0] = dirws[r * 4 + 0];
    out[1] = dirws[r * 4 + 1];
    out[2] = dirws[r * 4 + 2];
    const int j = nbr[r];
    const float* f = featws + (size_t)j * 16;
#pragma unroll
    for (int m = 0; m < 5; ++m) {
        float f0 = f[3 * m + 0], f1 = f[3 * m + 1], f2 = f[3 * m + 2];
#pragma unroll
        for (int i = 0; i < 3; ++i)
            out[3 + m * 3 + i] = f0 * V[0 * 3 + i] + f1 * V[1 * 3 + i] + f2 * V[2 * 3 + i];
    }
    _Float16* xp = X + r * 32;
#pragma unroll
    for (int i = 0; i < 18; ++i) xp[i] = (_Float16)out[i];
#pragma unroll
    for (int i = 18; i < 32; ++i) xp[i] = (_Float16)0.f;
}

// ---------------------------------------------------------------------------
// Kernel: pack weight matrix [Krows x Ncols] f32 into fp16 WMMA B-fragment
// layout: tile(kt,ct) holds 32 lanes x 8 dwords; lane n=lane&15 (col),
// lh=lane>>4, dword v packs (K, K+1) with K = kt*32 + (v<4 ? lh*8+2v : 16+lh*8+2(v-4)).
__global__ void __launch_bounds__(256) pack_b_kernel(const float* __restrict__ W,
                                                     int Krows, int Ncols, int ktiles,
                                                     _Float16* __restrict__ out) {
    const int idx = blockIdx.x * 256 + threadIdx.x;
    const int total = ktiles * (Ncols / 16) * 256;
    if (idx >= total) return;
    const int v    = idx & 7;
    const int lane = (idx >> 3) & 31;
    const int tile = idx >> 8;
    const int nct  = Ncols / 16;
    const int ct   = tile % nct;
    const int kt   = tile / nct;
    const int n    = ct * 16 + (lane & 15);
    const int lh   = lane >> 4;
    const int k    = kt * 32 + ((v < 4) ? (lh * 8 + 2 * v) : (16 + lh * 8 + 2 * (v - 4)));
    float f0 = (k     < Krows) ? W[(size_t)k * Ncols + n]       : 0.f;
    float f1 = (k + 1 < Krows) ? W[(size_t)(k + 1) * Ncols + n] : 0.f;
    out[(size_t)idx * 2 + 0] = (_Float16)f0;
    out[(size_t)idx * 2 + 1] = (_Float16)f1;
}

// ---------------------------------------------------------------------------
// Kernel 4: fused  relu(relu(X@W3+b3)@W4+b4)  +  per-point K-reduction.
// Block = 256 threads (8 waves), 80 rows = 4 points. Wave w owns 32 columns.
__global__ void __launch_bounds__(256) mlp_kernel(const _Float16* __restrict__ X,
                                                  const unsigned int* __restrict__ W3p,
                                                  const float* __restrict__ b3,
                                                  const unsigned int* __restrict__ W4p,
                                                  const float* __restrict__ b4,
                                                  float* __restrict__ point_feat) {
    __shared__ __align__(16) _Float16 Xs[80 * 32];
    __shared__ __align__(16) _Float16 H2s[80 * 256];
    __shared__ float accs[4 * 256];
    __shared__ float sb3[256], sb4[256];

    const int tid  = threadIdx.x;
    const int wave = tid >> 5;
    const int lane = tid & 31;
    const int m    = lane & 15;
    const int lh   = lane >> 4;
    const size_t r0 = (size_t)blockIdx.x * 80;

    {   // stage X tile: 80*32 halves = 320 u32x4
        const u32x4* src = reinterpret_cast<const u32x4*>(X + r0 * 32);
        u32x4* dst = reinterpret_cast<u32x4*>(Xs);
        for (int i = tid; i < 320; i += 256) dst[i] = src[i];
    }
    sb3[tid] = b3[tid];
    sb4[tid] = b4[tid];
    for (int i = tid; i < 1024; i += 256) accs[i] = 0.f;
    __syncthreads();

    const int colBase = wave * 32;

    // ---- GEMM1: [80x32] @ W3[32x256] -> relu -> H2s fp16 --------------------
    v8f c1[5][2];
#pragma unroll
    for (int rt = 0; rt < 5; ++rt) {
        v16h a = frag_a(Xs + (16 * rt + m) * 32, 0, lh);
#pragma unroll
        for (int cc = 0; cc < 2; ++cc) {
            const int ct = (colBase >> 4) + cc;
            v16h bf = frag_b(W3p + (size_t)ct * 256, lane);
            v8f c = {};
            c = __builtin_amdgcn_wmma_f32_16x16x32_f16(false, a, false, bf,
                                                       (short)0, c, false, false);
            c1[rt][cc] = c;
        }
    }
#pragma unroll
    for (int rt = 0; rt < 5; ++rt)
#pragma unroll
        for (int cc = 0; cc < 2; ++cc) {
            const int col = colBase + cc * 16 + m;
            const float bb = sb3[col];
#pragma unroll
            for (int rr = 0; rr < 8; ++rr) {
                const int rowl = 16 * rt + rr + 8 * lh;
                H2s[rowl * 256 + col] = (_Float16)fmaxf(c1[rt][cc][rr] + bb, 0.f);
            }
        }
    __syncthreads();

    // ---- GEMM2: [80x256] @ W4[256x256] -> relu -> per-point sum -------------
    v8f c2[5][2];
#pragma unroll
    for (int rt = 0; rt < 5; ++rt)
#pragma unroll
        for (int cc = 0; cc < 2; ++cc) { v8f z = {}; c2[rt][cc] = z; }

    for (int ks = 0; ks < 8; ++ks) {
#pragma unroll
        for (int rt = 0; rt < 5; ++rt) {
            v16h a = frag_a(H2s + (16 * rt + m) * 256, ks * 32, lh);
#pragma unroll
            for (int cc = 0; cc < 2; ++cc) {
                const int ct = (colBase >> 4) + cc;
                v16h bf = frag_b(W4p + (size_t)(ks * 16 + ct) * 256, lane);
                c2[rt][cc] = __builtin_amdgcn_wmma_f32_16x16x32_f16(
                    false, a, false, bf, (short)0, c2[rt][cc], false, false);
            }
        }
    }

#pragma unroll
    for (int rt = 0; rt < 5; ++rt)
#pragma unroll
        for (int cc = 0; cc < 2; ++cc) {
            const int col = colBase + cc * 16 + m;
            const float bb = sb4[col];
#pragma unroll
            for (int rr = 0; rr < 8; ++rr) {
                const int rowl = 16 * rt + rr + 8 * lh;   // 0..79
                const int pl   = rowl / KK;               // 0..3
                float v = fmaxf(c2[rt][cc][rr] + bb, 0.f);
                atomicAdd(&accs[pl * 256 + col], v);      // ds_add_f32
            }
        }
    __syncthreads();

    float* pf = point_feat + (size_t)blockIdx.x * 4 * 256;
    for (int i = tid; i < 1024; i += 256) pf[i] = accs[i];
}

// ---------------------------------------------------------------------------
// Kernel 5a: mean over P points per batch -> ys [16 x 256].
__global__ void __launch_bounds__(256) mean_kernel(const float* __restrict__ pf,
                                                   float* __restrict__ ys) {
    const int b = blockIdx.x, c = threadIdx.x;
    const float* p = pf + (size_t)b * PP * 256 + c;
    float s = 0.f;
    for (int i = 0; i < PP; ++i) s += p[(size_t)i * 256];
    ys[b * 256 + c] = s * (1.f / (float)PP);
}

// ---------------------------------------------------------------------------
// Kernel 5b: head: elu(ys@Wn1+bn1) -> log_softmax(. @Wn2 + bn2).
__global__ void __launch_bounds__(256) head_kernel(const float* __restrict__ ys,
                                                   const float* __restrict__ Wn1,
                                                   const float* __restrict__ bn1,
                                                   const float* __restrict__ Wn2,
                                                   const float* __restrict__ bn2,
                                                   float* __restrict__ out) {
    __shared__ float sy[BB * 256];
    __shared__ float sy1[BB * 256];
    __shared__ float sz[BB * NCLS];
    const int t = threadIdx.x;
    for (int i = t; i < BB * 256; i += 256) sy[i] = ys[i];
    __syncthreads();
    {
        const int c = t;
        const float bb = bn1[c];
        for (int b = 0; b < BB; ++b) {
            float acc = bb;
            for (int j = 0; j < 256; ++j) acc += sy[b * 256 + j] * Wn1[j * 256 + c];
            sy1[b * 256 + c] = (acc > 0.f) ? acc : (expf(acc) - 1.f);
        }
    }
    __syncthreads();
    for (int idx = t; idx < BB * NCLS; idx += 256) {
        const int b = idx / NCLS, o = idx % NCLS;
        float acc = bn2[o];
        for (int j = 0; j < 256; ++j) acc += sy1[b * 256 + j] * Wn2[j * NCLS + o];
        sz[b * NCLS + o] = acc;
    }
    __syncthreads();
    if (t < BB) {
        const int b = t;
        float mx = -1e30f;
        for (int o = 0; o < NCLS; ++o) mx = fmaxf(mx, sz[b * NCLS + o]);
        float s = 0.f;
        for (int o = 0; o < NCLS; ++o) s += expf(sz[b * NCLS + o] - mx);
        const float lse = mx + logf(s);
        for (int o = 0; o < NCLS; ++o) out[b * NCLS + o] = sz[b * NCLS + o] - lse;
    }
}

// ---------------------------------------------------------------------------
extern "C" void kernel_launch(void* const* d_in, const int* in_sizes, int n_in,
                              void* d_out, int out_size, void* d_ws, size_t ws_size,
                              hipStream_t stream) {
    (void)in_sizes; (void)n_in; (void)out_size; (void)ws_size;
    const float* pos = (const float*)d_in[0];
    const float* W1  = (const float*)d_in[1];
    const float* b1  = (const float*)d_in[2];
    const float* W2  = (const float*)d_in[3];
    const float* b2  = (const float*)d_in[4];
    const float* W3  = (const float*)d_in[5];
    const float* b3  = (const float*)d_in[6];
    const float* W4  = (const float*)d_in[7];
    const float* b4  = (const float*)d_in[8];
    const float* Wn1 = (const float*)d_in[9];
    const float* bn1 = (const float*)d_in[10];
    const float* Wn2 = (const float*)d_in[11];
    const float* bn2 = (const float*)d_in[12];
    float* out = (float*)d_out;

    char* ws = (char*)d_ws;
    size_t off = 0;
    auto carve = [&](size_t bytes) -> void* {
        void* p = ws + off;
        off = (off + bytes + 255) & ~(size_t)255;
        return p;
    };
    int*       nbr    = (int*)      carve((size_t)NPTS * KK * 4);
    float*     Vws    = (float*)    carve((size_t)NPTS * 9 * 4);
    float*     featws = (float*)    carve((size_t)NPTS * 16 * 4);
    float*     dirws  = (float*)    carve((size_t)RR * 4 * 4);
    _Float16*  Xcat   = (_Float16*) carve((size_t)RR * 32 * 2);
    _Float16*  W3p    = (_Float16*) carve((size_t)16 * 256 * 4);
    _Float16*  W4p    = (_Float16*) carve((size_t)128 * 256 * 4);
    float*     pfeat  = (float*)    carve((size_t)NPTS * 256 * 4);
    float*     ys     = (float*)    carve((size_t)BB * 256 * 4);

    knn_kernel   <<<BB * 4,       256, 0, stream>>>(pos, nbr);
    geom_kernel  <<<NPTS / 256,   256, 0, stream>>>(pos, nbr, W1, b1, W2, b2,
                                                    Vws, featws, dirws);
    concat_kernel<<<RR / 256,     256, 0, stream>>>(nbr, Vws, featws, dirws, Xcat);
    pack_b_kernel<<<16,           256, 0, stream>>>(W3, 18, 256, 1, W3p);
    pack_b_kernel<<<128,          256, 0, stream>>>(W4, 256, 256, 8, W4p);
    mlp_kernel   <<<NPTS / 4,     256, 0, stream>>>(Xcat,
                                                    (const unsigned int*)W3p, b3,
                                                    (const unsigned int*)W4p, b4,
                                                    pfeat);
    mean_kernel  <<<BB,           256, 0, stream>>>(pfeat, ys);
    head_kernel  <<<1,            256, 0, stream>>>(ys, Wn1, bn1, Wn2, bn2, out);
}
